// GAT_14370960572981
// MI455X (gfx1250) — compile-verified, matrix operand order checked by reading, not run on hardware
//
#include <hip/hip_runtime.h>
#include <math.h>

// ---------------- problem constants (from reference) ----------------
#define NN   60000
#define EEE  120000
#define GGG  2000
#define EMB  300
#define HHH  2
#define HD   600          // H * EMB
#define FEAT 512
#define FPD  1489
#define LLL  5
#define EP   (EEE + NN)   // edges incl. self loops
#define BN_EPS 1e-5f

typedef __attribute__((ext_vector_type(16))) _Float16 v16h;
typedef __attribute__((ext_vector_type(8)))  _Float16 v8h;
typedef __attribute__((ext_vector_type(8)))  float    v8f;

union F16Frag { v16h v; v8h h[2]; };

// =====================================================================
// WMMA GEMM:  C[M,N] = act( A[M,K] @ B[K,N] + bias[N] )
// Block: 128 threads (4 waves), tile 16(M) x 128(N), K stepped by 32.
// A panel (16x32) and B panel (32x128, transposed) staged in LDS as f16
// via pointer-incremented coalesced loads (partial tiles use clamped
// pointers + select, no divergent branches, no OOB); fragments read via
// ds_load_b128; each wave owns two 16x16 tiles -> two
// v_wmma_f32_16x16x32_f16 per K step.
// act: 0 = none, 1 = relu, 2 = softplus
// =====================================================================
__global__ void __launch_bounds__(128)
wmma_gemm_k(const float* __restrict__ A, int lda,
            const float* __restrict__ B, int ldb,
            const float* __restrict__ bias,
            float* __restrict__ C, int ldc,
            int M, int N, int K, int act) {
  __shared__ __align__(16) _Float16 As[16][40];    // 32 K + pad (16B-aligned slices)
  __shared__ __align__(16) _Float16 Bs[128][40];   // transposed: Bs[n][k]

  const int tid  = threadIdx.x;
  const int wave = tid >> 5;
  const int lane = tid & 31;
  const int half = lane >> 4;        // ISA: lanes 0-15 K-base 0, lanes 16-31 K-base 8
  const int lm   = lane & 15;
  const int kb   = half * 8;
  const int tm   = blockIdx.y * 16;
  const int tn0  = blockIdx.x * 128;

  v8f acc0 = {}, acc1 = {};

  for (int k0 = 0; k0 < K; k0 += 32) {
    // ---- prefetch next K panel (speculative, dropped if OOB) ----
    if (k0 + 32 < K) {
      __builtin_prefetch(&B[(size_t)(k0 + 32) * ldb + tn0 + tid], 0, 1);
    }

    // ---- stage A panel: 16 rows x 32 k, coalesced over k ----
    {
      const int kk = tid & 31;
      const int rr = tid >> 5;
      const bool full = (tm + 16 <= M) & (k0 + 32 <= K);
      if (full) {
        const float* ap = A + (size_t)(tm + rr) * lda + k0 + kk;
#pragma unroll
        for (int q = 0; q < 4; ++q) {
          As[q * 4 + rr][kk] = (_Float16)(*ap);
          ap += 4 * (size_t)lda;
        }
      } else {
        const int k  = k0 + kk;
        const int kc = k < K ? k : K - 1;            // clamped, always legal
#pragma unroll
        for (int q = 0; q < 4; ++q) {
          const int r  = q * 4 + rr;
          const int m  = tm + r;
          const int mc = m < M ? m : M - 1;
          const float v = A[(size_t)mc * lda + kc];  // unconditional load
          As[r][kk] = (_Float16)((m < M && k < K) ? v : 0.f);
        }
      }
    }

    // ---- stage B panel transposed: Bs[n][k], coalesced over n (=tid) ----
    {
      const bool full = (tn0 + 128 <= N) & (k0 + 32 <= K);
      const int n = tn0 + tid;
      if (full) {
        const float* bp = B + (size_t)k0 * ldb + n;
#pragma unroll 8
        for (int q = 0; q < 32; ++q) {
          Bs[tid][q] = (_Float16)(*bp);
          bp += ldb;
        }
      } else {
        const bool nv = (n < N);
        const int  nc = nv ? n : N - 1;              // clamped column
        const float* bp = B + (size_t)k0 * ldb + nc; // k0 < K here, so valid
#pragma unroll 8
        for (int q = 0; q < 32; ++q) {
          const bool kv = (k0 + q) < K;
          const float v = *bp;                       // pointer never leaves B
          Bs[tid][q] = (_Float16)((nv && kv) ? v : 0.f);
          bp += ((k0 + q + 1) < K) ? ldb : 0;        // clamped advance
        }
      }
    }
    __syncthreads();

    // ---- fragments from LDS (two ds_load_b128 each) ----
    F16Frag a, b0, b1;
    a.h[0]  = *(const v8h*)&As[lm][kb];
    a.h[1]  = *(const v8h*)&As[lm][kb + 16];
    const int nA = wave * 32 + lm;            // this wave's first sub-tile column
    b0.h[0] = *(const v8h*)&Bs[nA][kb];
    b0.h[1] = *(const v8h*)&Bs[nA][kb + 16];
    b1.h[0] = *(const v8h*)&Bs[nA + 16][kb];
    b1.h[1] = *(const v8h*)&Bs[nA + 16][kb + 16];

    acc0 = __builtin_amdgcn_wmma_f32_16x16x32_f16(false, a.v, false, b0.v,
                                                  (short)0, acc0, false, false);
    acc1 = __builtin_amdgcn_wmma_f32_16x16x32_f16(false, a.v, false, b1.v,
                                                  (short)0, acc1, false, false);
    __syncthreads();
  }

  // ---- epilogue: f32 C/D layout: VGPR v -> row tm + v + half*8 ----
#pragma unroll
  for (int v = 0; v < 8; ++v) {
    const int row  = tm + v + half * 8;
    const int col0 = tn0 + wave * 32 + lm;
    const int col1 = col0 + 16;
    if (row < M) {
      if (col0 < N) {
        float x = acc0[v] + (bias ? bias[col0] : 0.f);
        if (act == 1)      x = fmaxf(x, 0.f);
        else if (act == 2) x = (x > 20.f) ? x : log1pf(expf(x));
        C[(size_t)row * ldc + col0] = x;
      }
      if (col1 < N) {
        float x = acc1[v] + (bias ? bias[col1] : 0.f);
        if (act == 1)      x = fmaxf(x, 0.f);
        else if (act == 2) x = (x > 20.f) ? x : log1pf(expf(x));
        C[(size_t)row * ldc + col1] = x;
      }
    }
  }
}

// ---------------------------------------------------------------------
// small helpers
// ---------------------------------------------------------------------
__global__ void fill_k(float* __restrict__ p, long long n, float v) {
  long long t = (long long)blockIdx.x * blockDim.x + threadIdx.x;
  if (t < n) p[t] = v;
}

__device__ __forceinline__ void atomic_max_f32(float* addr, float v) {
  if (v >= 0.f) atomicMax((int*)addr, __float_as_int(v));
  else          atomicMin((unsigned int*)addr, (unsigned int)__float_as_int(v));
}

__device__ __forceinline__ void edge_decode(const int* __restrict__ ei,
                                            const int* __restrict__ ea,
                                            int e, int& i, int& j, int& a0, int& a1) {
  if (e < EEE) { i = ei[e]; j = ei[EEE + e]; a0 = ea[2 * e]; a1 = ea[2 * e + 1]; }
  else         { int s = e - EEE; i = s; j = s; a0 = 4; a1 = 0; }   // self loop
}

// h = x_emb1[node_x[:,0]] + x_emb2[node_x[:,1]]
__global__ void embed_k(const int* __restrict__ node_x,
                        const float* __restrict__ e1, const float* __restrict__ e2,
                        float* __restrict__ h) {
  long long t = (long long)blockIdx.x * blockDim.x + threadIdx.x;
  if (t >= (long long)NN * EMB) return;
  int n = (int)(t / EMB), d = (int)(t % EMB);
  h[t] = e1[(size_t)node_x[2 * n] * EMB + d] + e2[(size_t)node_x[2 * n + 1] * EMB + d];
}

// alpha[e,h] = leaky_relu( <x_i, att_i> + <x_j + ee, att_j>, 0.2 )
__global__ void edge_alpha_k(const float* __restrict__ hl,
                             const int* __restrict__ ei, const int* __restrict__ ea,
                             const float* __restrict__ ee1, const float* __restrict__ ee2,
                             const float* __restrict__ att,
                             float* __restrict__ alpha) {
  int t = blockIdx.x * blockDim.x + threadIdx.x;
  if (t >= EP * HHH) return;
  int e = t / HHH, hh = t % HHH;
  int i, j, a0, a1; edge_decode(ei, ea, e, i, j, a0, a1);
  const float* xi = hl + (size_t)i * HD + hh * EMB;
  const float* xj = hl + (size_t)j * HD + hh * EMB;
  const float* f1 = ee1 + (size_t)a0 * HD + hh * EMB;
  const float* f2 = ee2 + (size_t)a1 * HD + hh * EMB;
  const float* ai = att + (size_t)hh * 2 * EMB;
  const float* aj = ai + EMB;
  float s = 0.f;
  for (int d = 0; d < EMB; ++d)
    s = fmaf(xi[d], ai[d], fmaf(xj[d] + f1[d] + f2[d], aj[d], s));
  alpha[t] = (s > 0.f) ? s : 0.2f * s;
}

__global__ void segmax_k(const float* __restrict__ alpha,
                         const int* __restrict__ ei, const int* __restrict__ ea,
                         float* __restrict__ amax) {
  int t = blockIdx.x * blockDim.x + threadIdx.x;
  if (t >= EP * HHH) return;
  int e = t / HHH, hh = t % HHH;
  int i, j, a0, a1; edge_decode(ei, ea, e, i, j, a0, a1);
  atomic_max_f32(&amax[(size_t)i * HHH + hh], alpha[t]);
}

// ex = exp(alpha - amax[i]);  den[i] += ex   (alpha overwritten with ex)
__global__ void expden_k(float* __restrict__ alpha,
                         const int* __restrict__ ei, const int* __restrict__ ea,
                         const float* __restrict__ amax, float* __restrict__ den) {
  int t = blockIdx.x * blockDim.x + threadIdx.x;
  if (t >= EP * HHH) return;
  int e = t / HHH, hh = t % HHH;
  int i, j, a0, a1; edge_decode(ei, ea, e, i, j, a0, a1);
  float ex = expf(alpha[t] - amax[(size_t)i * HHH + hh]);
  alpha[t] = ex;
  atomicAdd(&den[(size_t)i * HHH + hh], ex);
}

// agg[i,d] += sum_h x_j[h,d] * ex[e,h]/(den[i,h]+1e-16)
__global__ void agg_k(const float* __restrict__ hl,
                      const int* __restrict__ ei, const int* __restrict__ ea,
                      const float* __restrict__ ee1, const float* __restrict__ ee2,
                      const float* __restrict__ ex, const float* __restrict__ den,
                      float* __restrict__ agg) {
  long long t = (long long)blockIdx.x * blockDim.x + threadIdx.x;
  if (t >= (long long)EP * EMB) return;
  int e = (int)(t / EMB), d = (int)(t % EMB);
  int i, j, a0, a1; edge_decode(ei, ea, e, i, j, a0, a1);
  float val = 0.f;
#pragma unroll
  for (int hh = 0; hh < HHH; ++hh) {
    float xj = hl[(size_t)j * HD + hh * EMB + d]
             + ee1[(size_t)a0 * HD + hh * EMB + d]
             + ee2[(size_t)a1 * HD + hh * EMB + d];
    float a = ex[(size_t)e * HHH + hh] / (den[(size_t)i * HHH + hh] + 1e-16f);
    val = fmaf(xj, a, val);
  }
  atomicAdd(&agg[(size_t)i * EMB + d], val);
}

// h = agg / H + bias
__global__ void headmean_k(const float* __restrict__ agg, const float* __restrict__ bias,
                           float* __restrict__ h) {
  long long t = (long long)blockIdx.x * blockDim.x + threadIdx.x;
  if (t >= (long long)NN * EMB) return;
  h[t] = agg[t] * (1.f / HHH) + bias[t % EMB];
}

// batch-norm statistics: one block per feature dim
__global__ void bn_stats_k(const float* __restrict__ h, float* __restrict__ mu,
                           float* __restrict__ var) {
  __shared__ float ls[256], lq[256];
  const int d = blockIdx.x;
  float s = 0.f, q = 0.f;
  for (int r = threadIdx.x; r < NN; r += blockDim.x) {
    float v = h[(size_t)r * EMB + d];
    s += v; q = fmaf(v, v, q);
  }
  ls[threadIdx.x] = s; lq[threadIdx.x] = q;
  __syncthreads();
  for (int st = 128; st > 0; st >>= 1) {
    if (threadIdx.x < st) { ls[threadIdx.x] += ls[threadIdx.x + st];
                            lq[threadIdx.x] += lq[threadIdx.x + st]; }
    __syncthreads();
  }
  if (threadIdx.x == 0) {
    float m = ls[0] / (float)NN;
    mu[d]  = m;
    var[d] = lq[0] / (float)NN - m * m;   // biased variance
  }
}

__global__ void bn_apply_k(float* __restrict__ h, const float* __restrict__ mu,
                           const float* __restrict__ var, const float* __restrict__ g,
                           const float* __restrict__ b, int relu) {
  long long t = (long long)blockIdx.x * blockDim.x + threadIdx.x;
  if (t >= (long long)NN * EMB) return;
  int d = (int)(t % EMB);
  float x = (h[t] - mu[d]) * rsqrtf(var[d] + BN_EPS) * g[d] + b[d];
  if (relu) x = fmaxf(x, 0.f);
  h[t] = x;
}

__global__ void pool_add_k(const float* __restrict__ h, const int* __restrict__ batch,
                           float* __restrict__ hg, float* __restrict__ cnt) {
  long long t = (long long)blockIdx.x * blockDim.x + threadIdx.x;
  if (t >= (long long)NN * EMB) return;
  int n = (int)(t / EMB), d = (int)(t % EMB);
  int g = batch[n];
  atomicAdd(&hg[(size_t)g * EMB + d], h[t]);
  if (d == 0) atomicAdd(&cnt[g], 1.f);
}

__global__ void pool_div_k(float* __restrict__ hg, const float* __restrict__ cnt) {
  int t = blockIdx.x * blockDim.x + threadIdx.x;
  if (t >= GGG * EMB) return;
  hg[t] /= fmaxf(cnt[t / EMB], 1.f);
}

// ---------------------------------------------------------------------
// host side
// ---------------------------------------------------------------------
static void gemm(hipStream_t s, const float* A, int lda, const float* B, int ldb,
                 const float* bias, float* C, int ldc, int M, int N, int K, int act) {
  dim3 grid((N + 127) / 128, (M + 15) / 16);
  wmma_gemm_k<<<grid, dim3(128), 0, s>>>(A, lda, B, ldb, bias, C, ldc, M, N, K, act);
}

#define L1D(kern, n, ...) \
  kern<<<(int)((((long long)(n)) + 255) / 256), 256, 0, stream>>>(__VA_ARGS__)

extern "C" void kernel_launch(void* const* d_in, const int* in_sizes, int n_in,
                              void* d_out, int out_size, void* d_ws, size_t ws_size,
                              hipStream_t stream) {
  // ---- inputs (setup_inputs order) ----
  const int*   node_x = (const int*)  d_in[0];
  const int*   ei     = (const int*)  d_in[1];   // [2,E] flat
  const int*   ea     = (const int*)  d_in[2];   // [E,2] flat
  const int*   batch  = (const int*)  d_in[3];
  const float* fp     = (const float*)d_in[4];
  const float* xe1    = (const float*)d_in[5];
  const float* xe2    = (const float*)d_in[6];
  const float* Wl     = (const float*)d_in[7];   // [L,EMB,HD]
  const float* bl     = (const float*)d_in[8];   // [L,HD]
  const float* attl   = (const float*)d_in[9];   // [L,H,2*EMB]
  const float* biasl  = (const float*)d_in[10];  // [L,EMB]
  const float* ee1l   = (const float*)d_in[11];  // [L,6,HD]
  const float* ee2l   = (const float*)d_in[12];  // [L,3,HD]
  const float* bng    = (const float*)d_in[13];
  const float* bnb    = (const float*)d_in[14];
  const float* fw     = (const float*)d_in[15];  // [EMB,FEAT]
  const float* fb     = (const float*)d_in[16];
  const float* f1w    = (const float*)d_in[17];  // [FPD,1024]
  const float* f1b    = (const float*)d_in[18];
  const float* f2w    = (const float*)d_in[19];  // [1024,512]
  const float* f2b    = (const float*)d_in[20];
  const float* f3w    = (const float*)d_in[21];  // [512,256]
  const float* f3b    = (const float*)d_in[22];
  const float* p1w    = (const float*)d_in[23];  // [768,512]
  const float* p1b    = (const float*)d_in[24];
  const float* p2w    = (const float*)d_in[25];  // [512,64]
  const float* p2b    = (const float*)d_in[26];
  const float* p3w    = (const float*)d_in[27];  // [64,2]
  const float* p3b    = (const float*)d_in[28];
  (void)in_sizes; (void)n_in; (void)out_size; (void)ws_size;

  // ---- workspace carve-out (floats) ----
  float* W = (float*)d_ws;
  size_t o = 0;
  float* h    = W + o; o += (size_t)NN * EMB;    // node features
  float* hl   = W + o; o += (size_t)NN * HD;     // transformed (2 heads)
  float* agg  = W + o; o += (size_t)NN * EMB;    // head-summed aggregation
  float* alp  = W + o; o += (size_t)EP * HHH;    // alpha / exp(alpha)
  float* amax = W + o; o += (size_t)NN * HHH;
  float* den  = W + o; o += (size_t)NN * HHH;
  float* hg   = W + o; o += (size_t)GGG * EMB;
  float* cnt  = W + o; o += (size_t)GGG;
  float* mu   = W + o; o += (size_t)EMB;
  float* var  = W + o; o += (size_t)EMB;
  float* fe1  = W + o; o += (size_t)GGG * 1024;
  float* fe2  = W + o; o += (size_t)GGG * 512;
  float* pb1  = W + o; o += (size_t)GGG * 512;
  float* pb2  = W + o; o += (size_t)GGG * 64;

  // ---- outputs: c [G,768] then p [G,2] ----
  float* cbuf = (float*)d_out;
  float* pbuf = cbuf + (size_t)GGG * (FEAT + FEAT / 2);
  const int LDC = FEAT + FEAT / 2;               // 768

  // ---- node embedding ----
  L1D(embed_k, (long long)NN * EMB, node_x, xe1, xe2, h);

  // ---- 5 GAT layers ----
  for (int l = 0; l < LLL; ++l) {
    const float* Wd  = Wl   + (size_t)l * EMB * HD;
    const float* bd  = bl   + (size_t)l * HD;
    const float* att = attl + (size_t)l * HHH * 2 * EMB;
    const float* bia = biasl+ (size_t)l * EMB;
    const float* e1  = ee1l + (size_t)l * 6 * HD;
    const float* e2  = ee2l + (size_t)l * 3 * HD;

    // hl = h @ W + b   (WMMA)
    gemm(stream, h, EMB, Wd, HD, bd, hl, HD, NN, HD, EMB, 0);

    // attention logits + softmax over incoming edges
    L1D(edge_alpha_k, EP * HHH, hl, ei, ea, e1, e2, att, alp);
    L1D(fill_k, (long long)NN * HHH, amax, (long long)NN * HHH, -1e30f);
    L1D(segmax_k, EP * HHH, alp, ei, ea, amax);
    L1D(fill_k, (long long)NN * HHH, den, (long long)NN * HHH, 0.f);
    L1D(expden_k, EP * HHH, alp, ei, ea, amax, den);

    // weighted aggregation (heads fused into one atomic)
    L1D(fill_k, (long long)NN * EMB, agg, (long long)NN * EMB, 0.f);
    L1D(agg_k, (long long)EP * EMB, hl, ei, ea, e1, e2, alp, den, agg);
    L1D(headmean_k, (long long)NN * EMB, agg, bia, h);

    // batch norm (+ relu except last layer)
    bn_stats_k<<<EMB, 256, 0, stream>>>(h, mu, var);
    L1D(bn_apply_k, (long long)NN * EMB, h, mu, var,
        bng + (size_t)l * EMB, bnb + (size_t)l * EMB, (l < LLL - 1) ? 1 : 0);
  }

  // ---- graph mean pooling ----
  L1D(fill_k, (long long)GGG * EMB, hg, (long long)GGG * EMB, 0.f);
  L1D(fill_k, (long long)GGG, cnt, (long long)GGG, 0.f);
  L1D(pool_add_k, (long long)NN * EMB, h, batch, hg, cnt);
  L1D(pool_div_k, GGG * EMB, hg, cnt);

  // ---- heads: c = [hg@fw+fb | relu-MLP(fp)] ----
  gemm(stream, hg, EMB, fw, FEAT, fb, cbuf, LDC, GGG, FEAT, EMB, 0);
  gemm(stream, fp, FPD, f1w, 1024, f1b, fe1, 1024, GGG, 1024, FPD, 1);
  gemm(stream, fe1, 1024, f2w, 512, f2b, fe2, 512, GGG, 512, 1024, 1);
  gemm(stream, fe2, 512, f3w, FEAT / 2, f3b, cbuf + FEAT, LDC, GGG, FEAT / 2, 512, 0);

  // ---- p = (softplus ∘ softplus) MLP on c ----
  gemm(stream, cbuf, LDC, p1w, FEAT, p1b, pb1, FEAT, GGG, FEAT, LDC, 2);
  gemm(stream, pb1, FEAT, p2w, FEAT / 8, p2b, pb2, FEAT / 8, GGG, FEAT / 8, FEAT, 2);
  gemm(stream, pb2, FEAT / 8, p3w, 2, p3b, pbuf, 2, GGG, 2, FEAT / 8, 0);
}
